// RNADiffuser_22153441312936
// MI455X (gfx1250) — compile-verified, matrix operand order checked by reading, not run on hardware
//
#include <hip/hip_runtime.h>
#include <hip/hip_bf16.h>
#include <math.h>

#define B_ 8
#define N_ 512
#define C_ 32
#define DEG_ 16
#define L_ 3
#define E_ (N_*(N_-1)/2)      // 130816
#define M_ (N_*DEG_)          // 8192
#define SLOPE_ 0.01f
#define TILES_PER_B (E_/16)   // 8176 (exact)

typedef __attribute__((ext_vector_type(16))) _Float16 v16h;
typedef __attribute__((ext_vector_type(8)))  _Float16 v8h;
typedef __attribute__((ext_vector_type(8)))  float    v8f;

__device__ __forceinline__ float lrelu_f(float x){ return x >= 0.0f ? x : SLOPE_*x; }
__device__ __forceinline__ v8f splat8(float x){ return (v8f){x,x,x,x,x,x,x,x}; }

__device__ __forceinline__ v8f wmma_f16(v16h a, v16h b, v8f c){
  // D = A(16x32 f16) * B(32x16 f16) + C(16x16 f32)
  return __builtin_amdgcn_wmma_f32_16x16x32_f16(false, a, false, b, (short)0, c, false, false);
}

__device__ __forceinline__ v16h loadBfrag(const _Float16* __restrict__ wf, int f, int l){
  // 32 contiguous bytes per lane, 32B aligned
  return *(const v16h*)(wf + f*512 + l*16);
}

__device__ __forceinline__ v16h cat8(v8h a, v8h b){
  return __builtin_shufflevector(a, b, 0,1,2,3,4,5,6,7,8,9,10,11,12,13,14,15);
}

// ---------------------------------------------------------------------------
// Prep kernels
// ---------------------------------------------------------------------------

// node = emb[tokens]  (B*N x C)
__global__ void k_embed(const float* __restrict__ emb, const int* __restrict__ tokens,
                        float* __restrict__ node){
  int t = blockIdx.x*blockDim.x + threadIdx.x;
  if (t >= B_*N_*C_) return;
  int n = t / C_, c = t % C_;
  node[t] = emb[tokens[n]*C_ + c];
}

// triu pair table: e -> (i0, i1)
__global__ void k_pairs(int2* __restrict__ pairs){
  int e = blockIdx.x*blockDim.x + threadIdx.x;
  if (e >= E_) return;
  float tn = 2.0f*N_ - 1.0f;
  float disc = tn*tn - 8.0f*(float)e;
  int a = (int)floorf((tn - sqrtf(fmaxf(disc, 0.0f))) * 0.5f);
  if (a < 0) a = 0; if (a > N_-2) a = N_-2;
  while (a > 0 && a*(2*N_-a-1)/2 > e) --a;
  while (a < N_-2 && (a+1)*(2*N_-(a+1)-1)/2 <= e) ++a;
  int start = a*(2*N_-a-1)/2;
  pairs[e] = make_int2(a, e - start + a + 1);
}

// per directed edge: flat pair index into (B*E)
__global__ void k_pidx(const int* __restrict__ esrc, const int* __restrict__ edst,
                       int* __restrict__ pidx){
  int t = blockIdx.x*blockDim.x + threadIdx.x;
  if (t >= B_*M_) return;
  int b = t / M_;
  int sg = esrc[t] + b*N_;
  int dg = edst[t] + b*N_;
  int a = min(sg, dg), bo = max(sg, dg);
  int gi = a / N_;
  int ai = a - gi*N_, bi = bo - gi*N_;
  pidx[t] = gi*E_ + ai*(2*N_ - ai - 3)/2 + bi - 1;
}

__global__ void k_zero(float* __restrict__ p, int n){
  int t = blockIdx.x*blockDim.x + threadIdx.x;
  if (t < n) p[t] = 0.0f;
}

// msg = relu(node[src] + edgefeat(pidx)); neigh[dst] += msg
__global__ void k_message(const float* __restrict__ node,
                          const int* __restrict__ esrc, const int* __restrict__ edst,
                          const int* __restrict__ pidx,
                          const float* __restrict__ d_sims, const float* __restrict__ edge_ex,
                          const float* __restrict__ eW, const float* __restrict__ eb,
                          float* __restrict__ neigh){
  int t = blockIdx.x*blockDim.x + threadIdx.x;
  if (t >= B_*M_) return;
  int b = t / M_;
  int sg = esrc[t] + b*N_;
  int dg = edst[t] + b*N_;
  int pe = pidx[t];
  float dv = d_sims[pe], ex = edge_ex[pe];
  float sq = sqrtf(fabsf(dv)), d2 = dv*dv;
  #pragma unroll 8
  for (int c = 0; c < C_; ++c){
    float ef = ex*eW[c] + dv*eW[32+c] + sq*eW[64+c] + d2*eW[96+c] + eb[c];
    float msg = node[sg*C_ + c] + ef;
    msg = fmaxf(msg, 0.0f);
    atomicAdd(&neigh[dg*C_ + c], msg);
  }
}

// per-node GIN MLP: [neigh|node](64) -> lrelu(W1) -> W2 -> lrelu -> W3
__global__ __launch_bounds__(64) void k_gin_mlp(
    const float* __restrict__ node, const float* __restrict__ neigh,
    const float* __restrict__ W1, const float* __restrict__ b1,
    const float* __restrict__ W2, const float* __restrict__ b2,
    const float* __restrict__ W3, const float* __restrict__ b3,
    float* __restrict__ node_out){
  __shared__ float hin[64], h1[64], h2[32];
  int nid = blockIdx.x, t = threadIdx.x;
  if (t < 32){ hin[t] = neigh[nid*C_ + t]; hin[32 + t] = node[nid*C_ + t]; }
  __syncthreads();
  {
    float s = b1[t];
    #pragma unroll 8
    for (int i = 0; i < 64; ++i) s += hin[i]*W1[i*64 + t];
    h1[t] = lrelu_f(s);
  }
  __syncthreads();
  if (t < 32){
    float s = b2[t];
    #pragma unroll 8
    for (int i = 0; i < 64; ++i) s += h1[i]*W2[i*32 + t];
    h2[t] = s;
  }
  __syncthreads();
  if (t < 32){
    float s = b3[t];
    #pragma unroll 8
    for (int i = 0; i < 32; ++i) s += lrelu_f(h2[i])*W3[i*32 + t];
    node_out[nid*C_ + t] = s;
  }
}

__global__ void k_tof16(const float* __restrict__ src, _Float16* __restrict__ dst, int n){
  int t = blockIdx.x*blockDim.x + threadIdx.x;
  if (t < n) dst[t] = (_Float16)src[t];
}

// Pack fW1/fW2/fW3/fW4 into per-lane WMMA B-fragments (f16).
// frag ids: 0..17 = fW1(kc=f/6, nt=f%6); 18..23 = fW2; 24..25 = fW3; 26 = fW4 (col0 only)
__global__ void k_prepack(const float* __restrict__ fW1, const float* __restrict__ fW2,
                          const float* __restrict__ fW3, const float* __restrict__ fW4,
                          _Float16* __restrict__ wf){
  int t = blockIdx.x*blockDim.x + threadIdx.x;
  if (t >= 27*512) return;
  int f = t / 512, r = t % 512;
  int l = r / 16, j = r % 16;
  int hf = l >> 4, nl = l & 15;
  int kl = (j < 8) ? (8*hf + j) : (16 + 8*hf + (j - 8));  // ISA 16-bit A/B K striping
  float v;
  if (f < 18)      { int kc = f/6,      nt = f%6;      v = fW1[(kc*32 + kl)*96 + nt*16 + nl]; }
  else if (f < 24) { int kc = (f-18)/2, nt = (f-18)%2; v = fW2[(kc*32 + kl)*32 + nt*16 + nl]; }
  else if (f < 26) { int nt = f-24;                    v = fW3[kl*32 + nt*16 + nl]; }
  else             { v = (nl == 0) ? fW4[kl] : 0.0f; }
  wf[t] = (_Float16)v;
}

// ---------------------------------------------------------------------------
// Hot kernel: per-pair scoring MLP. One wave32 per 16-pair tile, 27 WMMAs.
// ---------------------------------------------------------------------------
__global__ __launch_bounds__(32) void k_final(
    const float* __restrict__ d_sims, const float* __restrict__ edge_ex,
    const float* __restrict__ eW, const float* __restrict__ eb,
    const _Float16* __restrict__ nodef16, const int2* __restrict__ pairs,
    const _Float16* __restrict__ wf,
    const float* __restrict__ fb1, const float* __restrict__ fb2,
    const float* __restrict__ fb3, const float* __restrict__ fb4,
    float* __restrict__ out){
  __shared__ __align__(32) _Float16 smem[16*104];
  const int l  = threadIdx.x;
  const int m  = l & 15;        // row (layer input) / col (layer output) selector
  const int hf = l >> 4;        // K-half / M-half selector
  const int bb = blockIdx.x / TILES_PER_B;
  const int tb = blockIdx.x % TILES_PER_B;
  const int e_in = tb*16 + m;

  __builtin_prefetch(wf + l*64, 0, 3);   // global_prefetch_b8 over weight fragments

  int2 pr = pairs[e_in];
  const _Float16* rowL = nodef16 + (size_t)(bb*N_ + pr.x)*C_;
  const _Float16* rowR = nodef16 + (size_t)(bb*N_ + pr.y)*C_;
  v8h l0 = *(const v8h*)(rowL + 8*hf);
  v8h l1 = *(const v8h*)(rowL + 16 + 8*hf);
  v8h r0 = *(const v8h*)(rowR + 8*hf);
  v8h r1 = *(const v8h*)(rowR + 16 + 8*hf);

  const int fe = bb*E_ + e_in;
  float dv = d_sims[fe], ex = edge_ex[fe];
  float sq = sqrtf(fabsf(dv)), d2 = dv*dv;
  v8h e0, e1;
  #pragma unroll
  for (int j = 0; j < 8; ++j){
    int c0 = 8*hf + j;
    int c1 = 16 + 8*hf + j;
    float f0 = ex*eW[c0] + dv*eW[32+c0] + sq*eW[64+c0] + d2*eW[96+c0] + eb[c0];
    float f1 = ex*eW[c1] + dv*eW[32+c1] + sq*eW[64+c1] + d2*eW[96+c1] + eb[c1];
    e0[j] = (_Float16)f0;
    e1[j] = (_Float16)f1;
  }

  v16h a0 = cat8(l0, l1);   // cols  0..31 : lhs node
  v16h a1 = cat8(r0, r1);   // cols 32..63 : rhs node
  v16h a2 = cat8(e0, e1);   // cols 64..95 : edge features

  // ---- layer 1: 16x96 @ 96x96 ----
  #pragma unroll
  for (int nt = 0; nt < 6; ++nt){
    v8f acc = splat8(fb1[nt*16 + m]);
    acc = wmma_f16(a0, loadBfrag(wf, 0*6 + nt, l), acc);
    acc = wmma_f16(a1, loadBfrag(wf, 1*6 + nt, l), acc);
    acc = wmma_f16(a2, loadBfrag(wf, 2*6 + nt, l), acc);
    int col = nt*16 + m;
    #pragma unroll
    for (int r = 0; r < 8; ++r)
      smem[(8*hf + r)*104 + col] = (_Float16)lrelu_f(acc[r]);
  }
  __syncthreads();
  v16h h1f[3];
  #pragma unroll
  for (int kc = 0; kc < 3; ++kc){
    v8h g0 = *(const v8h*)(&smem[m*104 + kc*32 + 8*hf]);
    v8h g1 = *(const v8h*)(&smem[m*104 + kc*32 + 16 + 8*hf]);
    h1f[kc] = cat8(g0, g1);
  }
  __syncthreads();

  // ---- layer 2: 16x96 @ 96x32 (+lrelu) ----
  #pragma unroll
  for (int nt = 0; nt < 2; ++nt){
    v8f acc = splat8(fb2[nt*16 + m]);
    acc = wmma_f16(h1f[0], loadBfrag(wf, 18 + 0*2 + nt, l), acc);
    acc = wmma_f16(h1f[1], loadBfrag(wf, 18 + 1*2 + nt, l), acc);
    acc = wmma_f16(h1f[2], loadBfrag(wf, 18 + 2*2 + nt, l), acc);
    int col = nt*16 + m;
    #pragma unroll
    for (int r = 0; r < 8; ++r)
      smem[(8*hf + r)*40 + col] = (_Float16)lrelu_f(acc[r]);
  }
  __syncthreads();
  v16h a3;
  {
    v8h g0 = *(const v8h*)(&smem[m*40 + 8*hf]);
    v8h g1 = *(const v8h*)(&smem[m*40 + 16 + 8*hf]);
    a3 = cat8(g0, g1);
  }
  __syncthreads();

  // ---- layer 3: 16x32 @ 32x32 (+lrelu) ----
  #pragma unroll
  for (int nt = 0; nt < 2; ++nt){
    v8f acc = splat8(fb3[nt*16 + m]);
    acc = wmma_f16(a3, loadBfrag(wf, 24 + nt, l), acc);
    int col = nt*16 + m;
    #pragma unroll
    for (int r = 0; r < 8; ++r)
      smem[(8*hf + r)*40 + col] = (_Float16)lrelu_f(acc[r]);
  }
  __syncthreads();
  v16h a4;
  {
    v8h g0 = *(const v8h*)(&smem[m*40 + 8*hf]);
    v8h g1 = *(const v8h*)(&smem[m*40 + 16 + 8*hf]);
    a4 = cat8(g0, g1);
  }

  // ---- layer 4: 16x32 @ 32x16(col0=fW4) ----
  float bv = (m == 0) ? fb4[0] : 0.0f;
  v8f acc = splat8(bv);
  acc = wmma_f16(a4, loadBfrag(wf, 26, l), acc);
  if (m == 0){
    #pragma unroll
    for (int r = 0; r < 8; ++r)
      out[(size_t)bb*E_ + (size_t)tb*16 + 8*hf + r] = acc[r];
  }
}

// ---------------------------------------------------------------------------
extern "C" void kernel_launch(void* const* d_in, const int* in_sizes, int n_in,
                              void* d_out, int out_size, void* d_ws, size_t ws_size,
                              hipStream_t stream) {
  (void)in_sizes; (void)n_in; (void)out_size; (void)ws_size;
  const float* d_sims    = (const float*)d_in[0];
  const float* edge_ex   = (const float*)d_in[1];
  const float* emb       = (const float*)d_in[2];
  const float* edge_W    = (const float*)d_in[3];
  const float* edge_b    = (const float*)d_in[4];
  const float* gin_W1    = (const float*)d_in[5];
  const float* gin_b1    = (const float*)d_in[6];
  const float* gin_W2    = (const float*)d_in[7];
  const float* gin_b2    = (const float*)d_in[8];
  const float* gin_W3    = (const float*)d_in[9];
  const float* gin_b3    = (const float*)d_in[10];
  const float* fW1       = (const float*)d_in[11];
  const float* fb1       = (const float*)d_in[12];
  const float* fW2       = (const float*)d_in[13];
  const float* fb2       = (const float*)d_in[14];
  const float* fW3       = (const float*)d_in[15];
  const float* fb3       = (const float*)d_in[16];
  const float* fW4       = (const float*)d_in[17];
  const float* fb4       = (const float*)d_in[18];
  const int*   tokens    = (const int*)d_in[19];
  const int*   edge_src  = (const int*)d_in[20];
  const int*   edge_dst  = (const int*)d_in[21];
  float* out = (float*)d_out;

  char* ws = (char*)d_ws;
  int2*     pairs   = (int2*)(ws + 0);               // E_*8      ~ 1.0 MB
  float*    nodeA   = (float*)(ws + (1u<<20));       // 512 KB
  float*    nodeB   = (float*)(ws + (1u<<20) + 524288u);
  float*    neigh   = (float*)(ws + (1u<<20) + 1048576u);
  int*      pidx    = (int*)(ws + (1u<<20) + 1572864u);
  _Float16* nodef16 = (_Float16*)(ws + (1u<<20) + 1835008u);
  _Float16* wf      = (_Float16*)(ws + (1u<<20) + 2097152u);   // 27*512*2 = 27648 B

  k_pairs<<<(E_ + 255)/256, 256, 0, stream>>>(pairs);
  k_embed<<<(B_*N_*C_ + 255)/256, 256, 0, stream>>>(emb, tokens, nodeA);
  k_pidx<<<(B_*M_ + 255)/256, 256, 0, stream>>>(edge_src, edge_dst, pidx);
  k_prepack<<<(27*512 + 255)/256, 256, 0, stream>>>(fW1, fW2, fW3, fW4, wf);

  float* cur = nodeA;
  float* nxt = nodeB;
  for (int layer = 0; layer < L_; ++layer){
    k_zero<<<(B_*N_*C_ + 255)/256, 256, 0, stream>>>(neigh, B_*N_*C_);
    k_message<<<(B_*M_ + 255)/256, 256, 0, stream>>>(cur, edge_src, edge_dst, pidx,
                                                     d_sims, edge_ex, edge_W, edge_b, neigh);
    k_gin_mlp<<<B_*N_, 64, 0, stream>>>(cur, neigh,
                                        gin_W1 + layer*64*64, gin_b1 + layer*64,
                                        gin_W2 + layer*64*32, gin_b2 + layer*32,
                                        gin_W3 + layer*32*32, gin_b3 + layer*32,
                                        nxt);
    float* t = cur; cur = nxt; nxt = t;
  }
  // after 3 layers, final node state is in `cur`
  k_tof16<<<(B_*N_*C_ + 255)/256, 256, 0, stream>>>(cur, nodef16, B_*N_*C_);

  k_final<<<B_*TILES_PER_B, 32, 0, stream>>>(d_sims, edge_ex, edge_W, edge_b,
                                             nodef16, pairs, wf,
                                             fb1, fb2, fb3, fb4, out);
}
